// NasLSTM_48009144435423
// MI455X (gfx1250) — matrix operand order repaired and checked by reading, not compiled, not run on hardware
//
#include <hip/hip_runtime.h>

// ---------------- problem dims ----------------
constexpr int E  = 512;            // embedding dim
constexpr int H  = 1536;           // hidden dim
constexpr int G  = 4 * H;          // 6144 gate rows per cell
constexpr int TT = 48;             // controller steps
constexpr int S  = 32;             // choices per state
constexpr int VV = 128;            // embedding rows

constexpr int NBLK  = 96;
constexpr int NTHR  = 256;         // 8 wave32 per block

// ---------------- workspace layout (bytes) ----------------
constexpr size_t N_WIH1 = (size_t)G * E;
constexpr size_t N_WHH1 = (size_t)G * H;
constexpr size_t N_WIH2 = (size_t)G * H;
constexpr size_t N_WHH2 = (size_t)G * H;
constexpr size_t N_WOUT = (size_t)TT * S * H;
constexpr size_t N_EMB  = (size_t)VV * E;

constexpr size_t OFF_WIH1 = 0;
constexpr size_t OFF_WHH1 = OFF_WIH1 + N_WIH1 * 2;
constexpr size_t OFF_WIH2 = OFF_WHH1 + N_WHH1 * 2;
constexpr size_t OFF_WHH2 = OFF_WIH2 + N_WIH2 * 2;
constexpr size_t OFF_WOUT = OFF_WHH2 + N_WHH2 * 2;
constexpr size_t OFF_EMB  = OFF_WOUT + N_WOUT * 2;
constexpr size_t OFF_XBF  = OFF_EMB  + N_EMB  * 2;   // x:   512 bf16
constexpr size_t OFF_H1BF = OFF_XBF  + 4096;          // h1: 1536 bf16
constexpr size_t OFF_H2BF = OFF_H1BF + 4096;          // h2: 1536 bf16
constexpr size_t OFF_B1   = OFF_H2BF + 4096;          // b_ih1+b_hh1: 6144 f32
constexpr size_t OFF_B2   = OFF_B1   + 24576;         // b_ih2+b_hh2: 6144 f32
constexpr size_t OFF_G1   = OFF_B2   + 24576;         // gates1: 6144 f32
constexpr size_t OFF_G2   = OFF_G1   + 24576;         // gates2: 6144 f32
constexpr size_t OFF_C1   = OFF_G2   + 24576;         // c1: 1536 f32
constexpr size_t OFF_C2   = OFF_C1   + 8192;          // c2: 1536 f32
constexpr size_t OFF_BAR  = OFF_C2   + 8192;          // barrier cnt/gen

typedef __attribute__((ext_vector_type(16))) __bf16 v16bf;
typedef __attribute__((ext_vector_type(8)))  __bf16 v8bf;
typedef __attribute__((ext_vector_type(8)))  float  v8f;

// ---------------- grid-wide sense barrier (agent scope) ----------------
__device__ __forceinline__ void gbar(unsigned* cnt, unsigned* gen) {
  __threadfence();          // publish this thread's global stores to device scope
  __syncthreads();
  if (threadIdx.x == 0) {
    unsigned g = __hip_atomic_load(gen, __ATOMIC_ACQUIRE, __HIP_MEMORY_SCOPE_AGENT);
    unsigned a = __hip_atomic_fetch_add(cnt, 1u, __ATOMIC_ACQ_REL, __HIP_MEMORY_SCOPE_AGENT) + 1u;
    if (a == (unsigned)NBLK) {
      __hip_atomic_store(cnt, 0u, __ATOMIC_RELAXED, __HIP_MEMORY_SCOPE_AGENT);
      __hip_atomic_fetch_add(gen, 1u, __ATOMIC_RELEASE, __HIP_MEMORY_SCOPE_AGENT);
    } else {
      while (__hip_atomic_load(gen, __ATOMIC_ACQUIRE, __HIP_MEMORY_SCOPE_AGENT) == g)
        __builtin_amdgcn_s_sleep(1);
    }
  }
  __syncthreads();
  __threadfence();          // acquire: drop stale lines before reading others' data
}

// ---------------- single WMMA K-chunk: acc += Wtile[:,kb..kb+32) * vec[kb..kb+32) ----------------
// A (16x32 bf16): lane m=lane&15, elems 0..15 = K[voff..voff+16),
//                 voff = 16*(lane>=16)  (same contiguous K order as B)
// B (32x16 bf16): every column = vec segment -> per lane 16 contiguous K at kb+16*(lane>=16)
// D: all N identical; row = i + 8*(lane>=16) for acc[i]; read from lanes 0 and 16.
__device__ __forceinline__ void wmma_step(v8f& acc, const __bf16* __restrict__ wrow,
                                          const __bf16* __restrict__ vec,
                                          int kb, int koff, int voff) {
  v8bf alo = *(const v8bf*)(wrow + kb + voff);        // global_load_b128
  v8bf ahi = *(const v8bf*)(wrow + kb + voff + 8);    // global_load_b128
  v16bf a = __builtin_shufflevector(alo, ahi, 0,1,2,3,4,5,6,7,8,9,10,11,12,13,14,15);
  v16bf b = *(const v16bf*)(vec + kb + voff);         // broadcast within half-wave
  acc = __builtin_amdgcn_wmma_f32_16x16x32_bf16(false, a, false, b, (short)0, acc,
                                                false, false);
}

// 16-row GEMV over K=[k0,k1) with 4 independent accumulator chains to hide the
// bf16 XDL pipeline latency (WMMA->WMMA RAW needs ~5 slots per ISA 7.12.1).
__device__ __forceinline__ v8f wmma_gemv(v8f acc, const __bf16* __restrict__ Wtile, int ldk,
                                         const __bf16* __restrict__ vec, int k0, int k1, int lane) {
  const int m    = lane & 15;
  const int koff = (lane >> 4) << 3;   // 0 or 8
  const int voff = (lane >> 4) << 4;   // 0 or 16
  const __bf16* wrow = Wtile + (size_t)m * (size_t)ldk;
  v8f a0 = acc;
  v8f a1 = {0.f,0.f,0.f,0.f,0.f,0.f,0.f,0.f};
  v8f a2 = a1, a3 = a1;
  int kb = k0;
  for (; kb + 128 <= k1; kb += 128) {
    wmma_step(a0, wrow, vec, kb,      koff, voff);
    wmma_step(a1, wrow, vec, kb + 32, koff, voff);
    wmma_step(a2, wrow, vec, kb + 64, koff, voff);
    wmma_step(a3, wrow, vec, kb + 96, koff, voff);
  }
  for (; kb < k1; kb += 32) wmma_step(a0, wrow, vec, kb, koff, voff);
  return (a0 + a1) + (a2 + a3);
}

__device__ __forceinline__ float sigm(float x) { return 1.0f / (1.0f + __expf(-x)); }

// ---------------- prepass: f32 -> bf16 weights, bias sums, state init ----------------
__device__ __forceinline__ void cvt(__bf16* d, const float* s, size_t n, size_t g0, size_t gs) {
  for (size_t i = g0; i < n; i += gs) d[i] = (__bf16)s[i];
}

__global__ void nas_prep(char* __restrict__ ws, const int* __restrict__ input_id,
                         const float* __restrict__ emb,
                         const float* __restrict__ Wih1, const float* __restrict__ Whh1,
                         const float* __restrict__ bih1, const float* __restrict__ bhh1,
                         const float* __restrict__ Wih2, const float* __restrict__ Whh2,
                         const float* __restrict__ bih2, const float* __restrict__ bhh2,
                         const float* __restrict__ Wout) {
  const size_t g0 = (size_t)blockIdx.x * blockDim.x + threadIdx.x;
  const size_t gs = (size_t)gridDim.x * blockDim.x;

  cvt((__bf16*)(ws + OFF_WIH1), Wih1, N_WIH1, g0, gs);
  cvt((__bf16*)(ws + OFF_WHH1), Whh1, N_WHH1, g0, gs);
  cvt((__bf16*)(ws + OFF_WIH2), Wih2, N_WIH2, g0, gs);
  cvt((__bf16*)(ws + OFF_WHH2), Whh2, N_WHH2, g0, gs);
  cvt((__bf16*)(ws + OFF_WOUT), Wout, N_WOUT, g0, gs);
  cvt((__bf16*)(ws + OFF_EMB),  emb,  N_EMB,  g0, gs);

  float* b1 = (float*)(ws + OFF_B1);
  float* b2 = (float*)(ws + OFF_B2);
  float* g1 = (float*)(ws + OFF_G1);
  for (size_t i = g0; i < (size_t)G; i += gs) {
    float s1 = bih1[i] + bhh1[i];  b1[i] = s1;  g1[i] = s1;   // gates1 pre-seeded for step 0
    b2[i] = bih2[i] + bhh2[i];
  }
  float* c1 = (float*)(ws + OFF_C1);
  float* c2 = (float*)(ws + OFF_C2);
  __bf16* h1 = (__bf16*)(ws + OFF_H1BF);
  __bf16* h2 = (__bf16*)(ws + OFF_H2BF);
  for (size_t i = g0; i < (size_t)H; i += gs) {
    c1[i] = 0.0f; c2[i] = 0.0f;
    h1[i] = (__bf16)0.0f; h2[i] = (__bf16)0.0f;
  }
  __bf16* x = (__bf16*)(ws + OFF_XBF);
  const size_t id = (size_t)input_id[0];
  for (size_t i = g0; i < (size_t)E; i += gs) x[i] = (__bf16)emb[id * E + i];
  if (g0 == 0) {
    unsigned* bar = (unsigned*)(ws + OFF_BAR);
    bar[0] = 0u; bar[1] = 0u;
  }
}

// ---------------- persistent controller ----------------
__global__ __launch_bounds__(NTHR) void nas_ctrl(char* __restrict__ ws,
                                                 const float* __restrict__ b_out,
                                                 float* __restrict__ out) {
  const __bf16* wih1 = (const __bf16*)(ws + OFF_WIH1);
  const __bf16* whh1 = (const __bf16*)(ws + OFF_WHH1);
  const __bf16* wih2 = (const __bf16*)(ws + OFF_WIH2);
  const __bf16* whh2 = (const __bf16*)(ws + OFF_WHH2);
  const __bf16* wout = (const __bf16*)(ws + OFF_WOUT);
  const __bf16* embb = (const __bf16*)(ws + OFF_EMB);
  __bf16* xbf  = (__bf16*)(ws + OFF_XBF);
  __bf16* h1bf = (__bf16*)(ws + OFF_H1BF);
  __bf16* h2bf = (__bf16*)(ws + OFF_H2BF);
  const float* b1 = (const float*)(ws + OFF_B1);
  const float* b2 = (const float*)(ws + OFF_B2);
  float* g1 = (float*)(ws + OFF_G1);
  float* g2 = (float*)(ws + OFF_G2);
  float* c1 = (float*)(ws + OFF_C1);
  float* c2 = (float*)(ws + OFF_C2);
  unsigned* bar = (unsigned*)(ws + OFF_BAR);

  const int lane = threadIdx.x & 31;
  const int wid  = (blockIdx.x << 3) | (threadIdx.x >> 5);
  const int gtid = blockIdx.x * NTHR + threadIdx.x;
  const int tile = wid >> 1;          // 0..383 -> gate rows [tile*16, tile*16+16)
  const int half = wid & 1;           // K-split
  const int row0 = tile << 4;

  __shared__ float s_lg[S];

  for (int t = 0; t < TT; ++t) {
    // ---- P1: gates1 += Wih1@x + Whh1@h1 (bias already seeded) ----
    {
      v8f acc = {0.f,0.f,0.f,0.f,0.f,0.f,0.f,0.f};
      if (half == 0) {
        acc = wmma_gemv(acc, wih1 + (size_t)row0 * E, E, xbf,  0,   E, lane);
        acc = wmma_gemv(acc, whh1 + (size_t)row0 * H, H, h1bf, 0, 512, lane);
      } else {
        acc = wmma_gemv(acc, whh1 + (size_t)row0 * H, H, h1bf, 512, H, lane);
      }
      if ((lane & 15) == 0) {
        const int mb = row0 + ((lane >> 4) << 3);
#pragma unroll
        for (int i = 0; i < 8; ++i) unsafeAtomicAdd(&g1[mb + i], acc[i]);
      }
    }
    gbar(bar, bar + 1);

    // ---- P2: pointwise LSTM1; also seed gates2 with bias ----
    if (gtid < H) {
      const int i = gtid;
      const float gi = g1[i], gf = g1[H + i], gg = g1[2 * H + i], go = g1[3 * H + i];
      const float cn = sigm(gf) * c1[i] + sigm(gi) * tanhf(gg);
      c1[i] = cn;
      h1bf[i] = (__bf16)(sigm(go) * tanhf(cn));
    }
    if (gtid < G) g2[gtid] = b2[gtid];
    gbar(bar, bar + 1);

    // ---- P3: gates2 += Wih2@h1 + Whh2@h2 ----
    {
      v8f acc = {0.f,0.f,0.f,0.f,0.f,0.f,0.f,0.f};
      if (half == 0) acc = wmma_gemv(acc, wih2 + (size_t)row0 * H, H, h1bf, 0, H, lane);
      else           acc = wmma_gemv(acc, whh2 + (size_t)row0 * H, H, h2bf, 0, H, lane);
      if ((lane & 15) == 0) {
        const int mb = row0 + ((lane >> 4) << 3);
#pragma unroll
        for (int i = 0; i < 8; ++i) unsafeAtomicAdd(&g2[mb + i], acc[i]);
      }
    }
    gbar(bar, bar + 1);

    // ---- P4: pointwise LSTM2; seed gates1 with bias for next step ----
    if (gtid < H) {
      const int i = gtid;
      const float gi = g2[i], gf = g2[H + i], gg = g2[2 * H + i], go = g2[3 * H + i];
      const float cn = sigm(gf) * c2[i] + sigm(gi) * tanhf(gg);
      c2[i] = cn;
      h2bf[i] = (__bf16)(sigm(go) * tanhf(cn));
    }
    if (gtid < G) g1[gtid] = b1[gtid];
    gbar(bar, bar + 1);

    // ---- P5 (block 0): logits = Wout[t]@h2, log-softmax, argmax, x = emb[id] ----
    if (blockIdx.x == 0) {
      const int w = threadIdx.x >> 5;
      if (w < 2) {
        const int r0 = w << 4;
        v8f acc = {0.f,0.f,0.f,0.f,0.f,0.f,0.f,0.f};
        acc = wmma_gemv(acc, wout + ((size_t)t * S + r0) * H, H, h2bf, 0, H, lane);
        if ((lane & 15) == 0) {
          const int mb = r0 + ((lane >> 4) << 3);
#pragma unroll
          for (int i = 0; i < 8; ++i) s_lg[mb + i] = acc[i];   // LDS stage
        }
      }
      __syncthreads();
      if (threadIdx.x < S) {
        const int l = threadIdx.x;
        const float v = s_lg[l] + b_out[t * S + l];
        float m = v;
#pragma unroll
        for (int o = 16; o > 0; o >>= 1) m = fmaxf(m, __shfl_xor(m, o, 32));
        float sum = __expf(v - m);
#pragma unroll
        for (int o = 16; o > 0; o >>= 1) sum += __shfl_xor(sum, o, 32);
        out[t * S + l] = v - (m + __logf(sum));                // log-softmax row
        // argmax, first-max tie-break like jnp.argmax
        float bv = v; int bi = l;
#pragma unroll
        for (int o = 16; o > 0; o >>= 1) {
          const float ov = __shfl_xor(bv, o, 32);
          const int   oi = __shfl_xor(bi, o, 32);
          if (ov > bv || (ov == bv && oi < bi)) { bv = ov; bi = oi; }
        }
        const int emb_id = (t & 3) * S + bi;                   // (t % SIZE)*S + pred
        ((v16bf*)xbf)[l] = ((const v16bf*)(embb + (size_t)emb_id * E))[l]; // 32*16 = 512
      }
    }
    gbar(bar, bar + 1);
  }
}

extern "C" void kernel_launch(void* const* d_in, const int* in_sizes, int n_in,
                              void* d_out, int out_size, void* d_ws, size_t ws_size,
                              hipStream_t stream) {
  const int*   input_id = (const int*)  d_in[0];
  const float* emb      = (const float*)d_in[1];
  const float* Wih1     = (const float*)d_in[2];
  const float* Whh1     = (const float*)d_in[3];
  const float* bih1     = (const float*)d_in[4];
  const float* bhh1     = (const float*)d_in[5];
  const float* Wih2     = (const float*)d_in[6];
  const float* Whh2     = (const float*)d_in[7];
  const float* bih2     = (const float*)d_in[8];
  const float* bhh2     = (const float*)d_in[9];
  const float* Wout     = (const float*)d_in[10];
  const float* bout     = (const float*)d_in[11];
  char* ws = (char*)d_ws;

  nas_prep<<<1024, 256, 0, stream>>>(ws, input_id, emb, Wih1, Whh1, bih1, bhh1,
                                     Wih2, Whh2, bih2, bhh2, Wout);
  nas_ctrl<<<NBLK, NTHR, 0, stream>>>(ws, bout, (float*)d_out);
}
